// Int8Linear_45612552683883
// MI455X (gfx1250) — matrix operand order repaired and checked by reading, not compiled, not run on hardware
//
#include <hip/hip_runtime.h>
#include <cstdint>

// ---------- types ----------
typedef __attribute__((ext_vector_type(16))) _Float16 v16h;
typedef __attribute__((ext_vector_type(8)))  float    v8f;
typedef __attribute__((ext_vector_type(4)))  float    f32x4;
typedef __attribute__((ext_vector_type(4)))  uint32_t u32x4;

#define M_DIM 8192
#define N_DIM 4096
#define K_DIM 4096
#define BM 256              // block tile M
#define BN 128              // block tile N
#define BK 64               // block tile K
#define LDH 72              // BK + 8 halfs pad -> conflict-friendly ds_load_b128
#define NKI (K_DIM / BK)

union FragU { u32x4 q[2]; v16h h; };

static __device__ __forceinline__ uint32_t pkh(float a, float b) {
  auto p = __builtin_amdgcn_cvt_pkrtz(a, b);   // v_cvt_pk_f16_f32 (RTZ)
  return __builtin_bit_cast(uint32_t, p);
}

// ---------------- kernel 1: blockwise int8 -> f16 dequant ----------------
__global__ void __launch_bounds__(256)
dequant_w_kernel(const int8_t* __restrict__ qw,
                 const float* __restrict__ scales,
                 _Float16* __restrict__ w16) {
  const int t = blockIdx.x * blockDim.x + threadIdx.x;     // 2,097,152 threads
  const int64_t base = (int64_t)t * 8;
  const float s = scales[t >> 3];                          // block = base/64
  const int2 raw = *(const int2*)(qw + base);
  const int lo = raw.x, hi = raw.y;
  u32x4 o;
  o.x = pkh((float)(int8_t)(lo      ) * s, (float)(int8_t)(lo >> 8 ) * s);
  o.y = pkh((float)(int8_t)(lo >> 16) * s, (float)(int8_t)(lo >> 24) * s);
  o.z = pkh((float)(int8_t)(hi      ) * s, (float)(int8_t)(hi >> 8 ) * s);
  o.w = pkh((float)(int8_t)(hi >> 16) * s, (float)(int8_t)(hi >> 24) * s);
  *(u32x4*)(w16 + base) = o;
}

// ---------------- kernel 1b: x fp32 -> f16 (one-time pass) ----------------
__global__ void __launch_bounds__(256)
x_to_f16_kernel(const float* __restrict__ x, _Float16* __restrict__ x16) {
  const int t = blockIdx.x * blockDim.x + threadIdx.x;     // 4,194,304 threads
  const int64_t base = (int64_t)t * 8;
  const f32x4 lo = *(const f32x4*)(x + base);
  const f32x4 hi = *(const f32x4*)(x + base + 4);
  u32x4 o;
  o.x = pkh(lo.x, lo.y);
  o.y = pkh(lo.z, lo.w);
  o.z = pkh(hi.x, hi.y);
  o.w = pkh(hi.z, hi.w);
  *(u32x4*)(x16 + base) = o;
}

// ---------------- kernel 2: f16 WMMA GEMM, C = x * W'^T + bias ----------------
// AF16=true : A operand already f16 (pre-converted in ws)
// AF16=false: A operand fp32, converted during LDS staging
template <bool AF16>
__global__ void __launch_bounds__(256)
gemm_wmma(const void* __restrict__ aPtr,
          const _Float16* __restrict__ w16,
          const float* __restrict__ bias,
          float* __restrict__ out) {
  __shared__ alignas(16) _Float16 As[2][BM * LDH];   // 2 x 36,864 B
  __shared__ alignas(16) _Float16 Bs[2][BN * LDH];   // 2 x 18,432 B

  const int tid  = threadIdx.x;
  const int lane = tid & 31;
  const int wave = tid >> 5;
  const int wm   = (wave & 3) * 64;   // 4 waves along M -> 64 rows each
  const int wn   = (wave >> 2) * 64;  // 2 waves along N -> 64 cols each
  const int m_base = blockIdx.y * BM;
  const int n_base = blockIdx.x * BN;

  // A staging: one K-row (64 elems) per thread; B staging: half row per thread
  const int arow = tid;               // 0..255
  const int brow = tid >> 1;          // 0..127
  const int bcol = (tid & 1) * 32;    // 0 or 32 halfs

  const float*    gA32 = (const float*)aPtr    + (int64_t)(m_base + arow) * K_DIM;
  const _Float16* gA16 = (const _Float16*)aPtr + (int64_t)(m_base + arow) * K_DIM;
  const _Float16* gB   = w16 + (int64_t)(n_base + brow) * K_DIM + bcol;

  _Float16* const sA0 = &As[0][0] + arow * LDH;
  _Float16* const sB0 = &Bs[0][0] + brow * LDH + bcol;

  f32x4 av32[16];   // fp32 A slice (fallback path)
  u32x4 av16[8];    // f16 A slice  (fast path)
  u32x4 bv[4];      // f16 B slice

  auto load_stage = [&](int koff) {
    if constexpr (AF16) {
      const u32x4* pa = (const u32x4*)(gA16 + koff);
#pragma unroll
      for (int j = 0; j < 8; ++j) av16[j] = pa[j];
    } else {
      const f32x4* pa = (const f32x4*)(gA32 + koff);
#pragma unroll
      for (int j = 0; j < 16; ++j) av32[j] = pa[j];
    }
    const u32x4* pb = (const u32x4*)(gB + koff);
#pragma unroll
    for (int j = 0; j < 4; ++j) bv[j] = pb[j];
  };
  auto store_stage = [&](int buf) {
    _Float16* sa = sA0 + buf * (BM * LDH);
    if constexpr (AF16) {
#pragma unroll
      for (int j = 0; j < 8; ++j) *(u32x4*)(sa + j * 8) = av16[j];
    } else {
#pragma unroll
      for (int j = 0; j < 8; ++j) {
        u32x4 pk;
        pk.x = pkh(av32[2*j].x,   av32[2*j].y);
        pk.y = pkh(av32[2*j].z,   av32[2*j].w);
        pk.z = pkh(av32[2*j+1].x, av32[2*j+1].y);
        pk.w = pkh(av32[2*j+1].z, av32[2*j+1].w);
        *(u32x4*)(sa + j * 8) = pk;
      }
    }
    _Float16* sb = sB0 + buf * (BN * LDH);
#pragma unroll
    for (int j = 0; j < 4; ++j) *(u32x4*)(sb + j * 8) = bv[j];
  };

  v8f acc[4][4];
#pragma unroll
  for (int i = 0; i < 4; ++i)
#pragma unroll
    for (int j = 0; j < 4; ++j)
      acc[i][j] = (v8f){0.f, 0.f, 0.f, 0.f, 0.f, 0.f, 0.f, 0.f};

  // prologue: tile 0 -> LDS buf 0
  load_stage(0);
  store_stage(0);
  __syncthreads();

  const int mrow = lane & 15;
  for (int kt = 0; kt < NKI; ++kt) {
    const int buf = kt & 1;
    if (kt + 1 < NKI) load_stage((kt + 1) * BK);   // issue global loads early

    const _Float16* abase = &As[buf][0];
    const _Float16* bbase = &Bs[buf][0];
#pragma unroll
    for (int ks = 0; ks < BK; ks += 32) {
      FragU aF[4], bF[4];
      // A frag: lanes 0-15 -> K {ks..ks+7, ks+16..ks+23}; lanes 16-31 -> +8
      const int ak = ks + ((lane >> 4) << 3);
#pragma unroll
      for (int mt = 0; mt < 4; ++mt) {
        const _Float16* p = abase + (wm + mt * 16 + mrow) * LDH + ak;
        aF[mt].q[0] = *(const u32x4*)p;
        aF[mt].q[1] = *(const u32x4*)(p + 16);
      }
      // B frag: lane n holds column n; lanes 0-15 -> K {ks..ks+7, ks+16..ks+23}; lanes 16-31 -> +8
      const int bk = ks + ((lane >> 4) << 3);
#pragma unroll
      for (int nt = 0; nt < 4; ++nt) {
        const _Float16* p = bbase + (wn + nt * 16 + mrow) * LDH + bk;
        bF[nt].q[0] = *(const u32x4*)p;
        bF[nt].q[1] = *(const u32x4*)(p + 16);
      }
#pragma unroll
      for (int mt = 0; mt < 4; ++mt)
#pragma unroll
        for (int nt = 0; nt < 4; ++nt)
          acc[mt][nt] = __builtin_amdgcn_wmma_f32_16x16x32_f16(
              false, aF[mt].h, false, bF[nt].h, (short)0, acc[mt][nt],
              false, false);
    }
    if (kt + 1 < NKI) store_stage((kt + 1) & 1);
    __syncthreads();
  }

  // epilogue: C/D layout -> VGPR r holds M=r (lanes 0-15) / M=r+8 (lanes 16-31)
  const int rbase = m_base + wm + ((lane >> 4) << 3);
#pragma unroll
  for (int nt = 0; nt < 4; ++nt) {
    const int col = n_base + wn + nt * 16 + mrow;
    const float bb = bias[col];
#pragma unroll
    for (int mt = 0; mt < 4; ++mt) {
#pragma unroll
      for (int r = 0; r < 8; ++r) {
        out[(int64_t)(rbase + mt * 16 + r) * N_DIM + col] = acc[mt][nt][r] + bb;
      }
    }
  }
}

extern "C" void kernel_launch(void* const* d_in, const int* in_sizes, int n_in,
                              void* d_out, int out_size, void* d_ws, size_t ws_size,
                              hipStream_t stream) {
  (void)in_sizes; (void)n_in; (void)out_size;
  const float*  x      = (const float*)d_in[0];
  const int8_t* qw     = (const int8_t*)d_in[1];
  const float*  scales = (const float*)d_in[2];
  const float*  bias   = (const float*)d_in[3];
  float*        out    = (float*)d_out;

  const size_t W16_BYTES = (size_t)N_DIM * K_DIM * 2;          // 32 MB
  const size_t X16_BYTES = (size_t)M_DIM * K_DIM * 2;          // 64 MB
  _Float16* w16 = (_Float16*)d_ws;
  _Float16* x16 = (_Float16*)((char*)d_ws + W16_BYTES);

  // 16,777,216 weights / 8 per thread / 256 per block = 8192 blocks
  dequant_w_kernel<<<8192, 256, 0, stream>>>(qw, scales, w16);

  dim3 grid(N_DIM / BN, M_DIM / BM);                           // (32, 32)
  if (ws_size >= W16_BYTES + X16_BYTES) {
    // fast path: pre-convert x to f16 once, GEMM streams f16 for both operands
    x_to_f16_kernel<<<16384, 256, 0, stream>>>(x, x16);
    gemm_wmma<true><<<grid, 256, 0, stream>>>(x16, w16, bias, out);
  } else {
    // fallback: convert x fp32->f16 during LDS staging
    gemm_wmma<false><<<grid, 256, 0, stream>>>(x, w16, bias, out);
  }
}
